// LigerFusedLinearGRPOLossVariant1_28690381537810
// MI455X (gfx1250) — compile-verified, matrix operand order checked by reading, not run on hardware
//
#include <hip/hip_runtime.h>
#include <hip/hip_bf16.h>
#include <math.h>

// Problem constants (from reference setup_inputs)
#define BDIM     8
#define TDIM     512
#define BT       4096          // B*T token rows
#define HDIM     1024          // hidden
#define VDIM     32000         // vocab
#define NSPLIT   10            // vocab splits per row-block
#define CTILES   2000          // VDIM/16 column tiles
#define TILES_PER_SPLIT 200    // CTILES / NSPLIT (even -> pairs of tiles)
#define MPANEL   128           // rows staged in LDS per workgroup (8 waves x 16)
#define LDS_STRIDE (HDIM + 8)  // bf16 elements; +8 kills DS bank conflicts

typedef __attribute__((ext_vector_type(16))) __bf16 v16bf;
typedef __attribute__((ext_vector_type(8)))  float  v8f;

__device__ __forceinline__ unsigned short f2bf(float f) {
    unsigned int x = __float_as_uint(f);
    unsigned int r = x + 0x7FFFu + ((x >> 16) & 1u);   // round-to-nearest-even
    return (unsigned short)(r >> 16);
}

// 16x32 bf16 WMMA operand fragment: elements 0..7 = K kb+0..7, 8..15 = K kb+16..23
struct FragBF {
    union { v16bf v; uint4 q[2]; };
};
__device__ __forceinline__ void load_frag(FragBF& f, const unsigned short* p) {
    f.q[0] = *(const uint4*)(p);        // 16 bytes: K + 0..7
    f.q[1] = *(const uint4*)(p + 16);   // 16 bytes: K + 16..23
}

// ---------------------------------------------------------------------------
// Kernel 1: fp32 weight -> bf16 (64 MB result: resident in 192 MB L2)
// ---------------------------------------------------------------------------
__global__ void convert_w_bf16(const float* __restrict__ w,
                               unsigned short* __restrict__ wbf, int n4) {
    int i = blockIdx.x * blockDim.x + threadIdx.x;
    if (i < n4) {
        float4 v = ((const float4*)w)[i];
        uint2 o;
        o.x = (unsigned)f2bf(v.x) | ((unsigned)f2bf(v.y) << 16);
        o.y = (unsigned)f2bf(v.z) | ((unsigned)f2bf(v.w) << 16);
        ((uint2*)wbf)[i] = o;
    }
}

// ---------------------------------------------------------------------------
// Kernel 2: fused GEMM + online logsumexp + selected-logit gather.
// grid.x = (BT/MPANEL) * NSPLIT, block = 256 (8 waves), dyn LDS = MPANEL panel
// Per wave: one 16-row block, TWO 16-col tiles at a time (independent WMMA
// accumulator chains), explicit double-buffered K loop.
// ---------------------------------------------------------------------------
__global__ void __launch_bounds__(256)
gemm_lse(const float* __restrict__ x,
         const unsigned short* __restrict__ wbf,
         const int* __restrict__ ids,
         float* __restrict__ pmax,
         float* __restrict__ psum,
         float* __restrict__ psel) {
    extern __shared__ unsigned short ldsA[];   // MPANEL * LDS_STRIDE bf16

    const int wgM   = blockIdx.x / NSPLIT;
    const int split = blockIdx.x % NSPLIT;

    // Stage + convert activation panel: 128 rows x 1024 fp32 -> bf16 in LDS
    {
        const float4* xp = (const float4*)(x + (size_t)wgM * MPANEL * HDIM);
        for (int i = threadIdx.x; i < MPANEL * HDIM / 4; i += blockDim.x) {
            int row  = i >> 8;          // 256 float4 per row
            int cpos = i & 255;
            float4 v = xp[i];
            uint2 o;
            o.x = (unsigned)f2bf(v.x) | ((unsigned)f2bf(v.y) << 16);
            o.y = (unsigned)f2bf(v.z) | ((unsigned)f2bf(v.w) << 16);
            *(uint2*)(ldsA + row * LDS_STRIDE + cpos * 4) = o;
        }
    }
    __syncthreads();

    const int lane   = threadIdx.x & 31;
    const int wave   = threadIdx.x >> 5;
    const int laneLo = lane & 15;
    const int laneHi = lane >> 4;          // 0 or 1
    const int kb     = laneHi * 8;         // A/B fragment K sub-offset

    // A fragment source: lanes 0-15 and 16-31 both address row = laneLo
    const unsigned short* aRow =
        ldsA + (wave * 16 + laneLo) * LDS_STRIDE + kb;

    // This lane's 8 C rows: M = r + 8*laneHi  (C layout: VGPR r, lane = N)
    const int gRowBase = wgM * MPANEL + wave * 16 + laneHi * 8;
    int selId[8];
#pragma unroll
    for (int r = 0; r < 8; ++r) selId[r] = ids[gRowBase + r];

    float runMax[8], runSum[8], selV[8];
#pragma unroll
    for (int r = 0; r < 8; ++r) {
        runMax[r] = -__builtin_inff();
        runSum[r] = 0.f;
        selV[r]   = -__builtin_inff();
    }

    const int cstart = split * TILES_PER_SPLIT * 16;

    for (int t = 0; t < TILES_PER_SPLIT; t += 2) {
        const int cbase0 = cstart + t * 16;
        const int cbase1 = cbase0 + 16;
        const unsigned short* bRow0 =
            wbf + (size_t)(cbase0 + laneLo) * HDIM + kb;
        const unsigned short* bRow1 = bRow0 + 16 * HDIM;
        if (t + 2 < TILES_PER_SPLIT) {          // pull next tile pair into L2/L0
            __builtin_prefetch(bRow0 + 32 * HDIM, 0, 1);
            __builtin_prefetch(bRow1 + 32 * HDIM, 0, 1);
        }

        v8f c0 = {0.f, 0.f, 0.f, 0.f, 0.f, 0.f, 0.f, 0.f};
        v8f c1 = {0.f, 0.f, 0.f, 0.f, 0.f, 0.f, 0.f, 0.f};

        // ---- double-buffered K loop: load (k+32) before consuming (k) ----
        FragBF aC, aN, b0C, b0N, b1C, b1N;
        load_frag(aC,  aRow);
        load_frag(b0C, bRow0);
        load_frag(b1C, bRow1);
#pragma unroll 4
        for (int k0 = 0; k0 < HDIM; k0 += 64) {
            load_frag(aN,  aRow  + k0 + 32);
            load_frag(b0N, bRow0 + k0 + 32);
            load_frag(b1N, bRow1 + k0 + 32);
            c0 = __builtin_amdgcn_wmma_f32_16x16x32_bf16(
                     false, aC.v, false, b0C.v, (short)0, c0, false, false);
            c1 = __builtin_amdgcn_wmma_f32_16x16x32_bf16(
                     false, aC.v, false, b1C.v, (short)0, c1, false, false);
            if (k0 + 64 < HDIM) {
                load_frag(aC,  aRow  + k0 + 64);
                load_frag(b0C, bRow0 + k0 + 64);
                load_frag(b1C, bRow1 + k0 + 64);
            }
            c0 = __builtin_amdgcn_wmma_f32_16x16x32_bf16(
                     false, aN.v, false, b0N.v, (short)0, c0, false, false);
            c1 = __builtin_amdgcn_wmma_f32_16x16x32_bf16(
                     false, aN.v, false, b1N.v, (short)0, c1, false, false);
        }

        // ---- online logsumexp update: 2 logits (one per tile) per C row ----
        const int col0 = cbase0 + laneLo;
        const int col1 = cbase1 + laneLo;
#pragma unroll
        for (int r = 0; r < 8; ++r) {
            float l0 = c0[r];            // TEMPERATURE == 1
            float nm = fmaxf(runMax[r], l0);
            runSum[r] = runSum[r] * __expf(runMax[r] - nm) + __expf(l0 - nm);
            runMax[r] = nm;
            selV[r]   = (col0 == selId[r]) ? l0 : selV[r];

            float l1 = c1[r];
            nm = fmaxf(runMax[r], l1);
            runSum[r] = runSum[r] * __expf(runMax[r] - nm) + __expf(l1 - nm);
            runMax[r] = nm;
            selV[r]   = (col1 == selId[r]) ? l1 : selV[r];
        }
    }

    // Reduce (max,sumexp,sel) across the 16 lanes that share each row.
    // xor masks 1,2,4,8 stay inside each 16-lane half of the wave32.
#pragma unroll
    for (int r = 0; r < 8; ++r) {
        float m = runMax[r], s = runSum[r], v = selV[r];
#pragma unroll
        for (int off = 1; off < 16; off <<= 1) {
            float m2 = __shfl_xor(m, off, 32);
            float s2 = __shfl_xor(s, off, 32);
            float v2 = __shfl_xor(v, off, 32);
            float nm = fmaxf(m, m2);
            s = s * __expf(m - nm) + s2 * __expf(m2 - nm);
            m = nm;
            v = fmaxf(v, v2);
        }
        runMax[r] = m; runSum[r] = s; selV[r] = v;
    }

    if (laneLo == 0) {      // lanes 0 and 16 each own 8 rows
#pragma unroll
        for (int r = 0; r < 8; ++r) {
            int idx = (gRowBase + r) * NSPLIT + split;
            pmax[idx] = runMax[r];
            psum[idx] = runSum[r];
            psel[idx] = selV[r];
        }
    }
}

// ---------------------------------------------------------------------------
// Kernel 3: merge NSPLIT partial logsumexps -> per-token logp
// ---------------------------------------------------------------------------
__global__ void row_reduce(const float* __restrict__ pmax,
                           const float* __restrict__ psum,
                           const float* __restrict__ psel,
                           float* __restrict__ logp, int nrows) {
    int r = blockIdx.x * blockDim.x + threadIdx.x;
    if (r >= nrows) return;
    float m = -__builtin_inff();
    for (int s = 0; s < NSPLIT; ++s) m = fmaxf(m, pmax[r * NSPLIT + s]);
    float sum = 0.f, sv = -__builtin_inff();
    for (int s = 0; s < NSPLIT; ++s) {
        sum += psum[r * NSPLIT + s] * __expf(pmax[r * NSPLIT + s] - m);
        sv = fmaxf(sv, psel[r * NSPLIT + s]);
    }
    logp[r] = sv - (m + __logf(sum));
}

// ---------------------------------------------------------------------------
// Kernel 4: percentile threshold (rank selection) + GRPO loss scalars
// ---------------------------------------------------------------------------
__global__ void __launch_bounds__(1024)
finalize(const float* __restrict__ logp_pol,
         const float* __restrict__ logp_ref,
         const float* __restrict__ att,
         const float* __restrict__ adv,
         float* __restrict__ out) {
    __shared__ float sRef[BT];
    __shared__ float red0[1024], red1[1024], red2[1024], red3[1024];
    __shared__ float thr;

    const int t = threadIdx.x;
    for (int i = t; i < BT; i += 1024) sRef[i] = logp_ref[i];
    __syncthreads();

    const int K = 819;   // max(1, int(4096 * 0.2)) -> sort(flat)[K-1]
    for (int i = t; i < BT; i += 1024) {
        float v = sRef[i];
        int lt = 0, le = 0;
        for (int j = 0; j < BT; ++j) {
            float u = sRef[j];
            lt += (u < v);
            le += (u <= v);
        }
        if (lt < K && le >= K) thr = v;   // unique value (ties write same v)
    }
    __syncthreads();
    const float thv = thr;

    float sl = 0.f, sk = 0.f, sa = 0.f, sc = 0.f;
    for (int i = t; i < BT; i += 1024) {
        float lp = logp_pol[i];
        float lr = sRef[i];
        float a  = att[i];
        float ad = adv[i >> 9];                 // b = i / T, T = 512
        float coef1 = __expf(lp - lp);          // old == stop_grad(pol) -> 1
        float coef2 = fminf(fmaxf(coef1, 0.8f), 1.2f);
        float ptl   = -fminf(coef1 * ad, coef2 * ad);
        float mask  = (lr <= thv) ? 1.f : 0.f;
        float d     = lr - lp;
        float k3    = __expf(d) - d - 1.f;
        float kl    = mask * k3 * 5.0f;         // 1/percentile
        ptl += 0.04f * kl;                      // BETA
        float clip = (((coef1 < 0.8f) && (ad < 0.f)) ||
                      ((coef1 > 1.2f) && (ad > 0.f))) ? 1.f : 0.f;
        sl += ptl * a; sk += kl * a; sa += a; sc += clip * a;
    }
    red0[t] = sl; red1[t] = sk; red2[t] = sa; red3[t] = sc;
    __syncthreads();
    for (int o = 512; o > 0; o >>= 1) {
        if (t < o) {
            red0[t] += red0[t + o]; red1[t] += red1[t + o];
            red2[t] += red2[t + o]; red3[t] += red3[t + o];
        }
        __syncthreads();
    }
    if (t == 0) {
        float norm = fmaxf(red2[0], 1.f);
        out[0] = red0[0] / norm;   // loss
        out[1] = red1[0] / norm;   // kl metric
        out[2] = red3[0] / norm;   // clip ratio
    }
}

// ---------------------------------------------------------------------------
extern "C" void kernel_launch(void* const* d_in, const int* in_sizes, int n_in,
                              void* d_out, int out_size, void* d_ws, size_t ws_size,
                              hipStream_t stream) {
    const float* x_pol = (const float*)d_in[0];
    const float* w_pol = (const float*)d_in[1];
    const int*   ids   = (const int*)d_in[2];
    const float* att   = (const float*)d_in[3];
    const float* adv   = (const float*)d_in[4];
    const float* x_ref = (const float*)d_in[5];
    const float* w_ref = (const float*)d_in[6];
    float* out = (float*)d_out;

    // Workspace layout (~67 MB): one bf16 W buffer reused for policy then ref.
    char* ws = (char*)d_ws;
    size_t off = 0;
    unsigned short* wbf = (unsigned short*)(ws + off); off += (size_t)VDIM * HDIM * 2;
    float* pmaxP = (float*)(ws + off); off += (size_t)BT * NSPLIT * 4;
    float* psumP = (float*)(ws + off); off += (size_t)BT * NSPLIT * 4;
    float* pselP = (float*)(ws + off); off += (size_t)BT * NSPLIT * 4;
    float* pmaxR = (float*)(ws + off); off += (size_t)BT * NSPLIT * 4;
    float* psumR = (float*)(ws + off); off += (size_t)BT * NSPLIT * 4;
    float* pselR = (float*)(ws + off); off += (size_t)BT * NSPLIT * 4;
    float* lpPol = (float*)(ws + off); off += (size_t)BT * 4;
    float* lpRef = (float*)(ws + off); off += (size_t)BT * 4;

    const int n4 = VDIM * HDIM / 4;
    const int convBlocks = (n4 + 255) / 256;
    const int gemmGrid = (BT / MPANEL) * NSPLIT;     // 32 * 10 = 320 workgroups
    const size_t gemmLds = (size_t)MPANEL * LDS_STRIDE * sizeof(unsigned short);

    // Policy pass
    convert_w_bf16<<<convBlocks, 256, 0, stream>>>(w_pol, wbf, n4);
    gemm_lse<<<gemmGrid, 256, gemmLds, stream>>>(x_pol, wbf, ids, pmaxP, psumP, pselP);
    // Reference pass (reuses wbf; stream order serializes vs. policy GEMM)
    convert_w_bf16<<<convBlocks, 256, 0, stream>>>(w_ref, wbf, n4);
    gemm_lse<<<gemmGrid, 256, gemmLds, stream>>>(x_ref, wbf, ids, pmaxR, psumR, pselR);

    row_reduce<<<BT / 256, 256, 0, stream>>>(pmaxP, psumP, pselP, lpPol, BT);
    row_reduce<<<BT / 256, 256, 0, stream>>>(pmaxR, psumR, pselR, lpRef, BT);

    finalize<<<1, 1024, 0, stream>>>(lpPol, lpRef, att, adv, out);
}
